// MultiTaskMaskedTokenModeling_11519102288297
// MI455X (gfx1250) — matrix operand order repaired and checked by reading, not compiled, not run on hardware
//
#include <hip/hip_runtime.h>
#include <hip/hip_bf16.h>

// ---------------------------------------------------------------------------
// MultiTaskMaskedTokenModeling forward pass for gfx1250 (MI455X).
// All dense math runs through v_wmma_f32_16x16x32_bf16 (bf16 in, fp32 acc).
// GEMM staging uses gfx1250 async global->LDS loads (ASYNCcnt) when the
// toolchain exposes them, with a sync fallback.
// ---------------------------------------------------------------------------

typedef __attribute__((ext_vector_type(16))) __bf16 v16bf;
typedef __attribute__((ext_vector_type(8)))  __bf16 v8bf;
typedef __attribute__((ext_vector_type(8)))  float  v8f;
typedef int vint4 __attribute__((vector_size(16)));

#ifdef __HIP_DEVICE_COMPILE__
#if __has_builtin(__builtin_amdgcn_global_load_async_to_lds_b128) && \
    __has_builtin(__builtin_amdgcn_s_wait_asynccnt)
#define USE_ASYNC_LDS 1
#endif
#endif
#ifndef USE_ASYNC_LDS
#define USE_ASYNC_LDS 0
#endif

__device__ __forceinline__ __bf16 f2bf(float f) {
  union { float f; unsigned u; } x; x.f = f;
  unsigned r = x.u + 0x7FFFu + ((x.u >> 16) & 1u);
  union { unsigned short s; __bf16 b; } y; y.s = (unsigned short)(r >> 16);
  return y.b;
}
__device__ __forceinline__ float bf2f(__bf16 b) {
  union { __bf16 b; unsigned short s; } x; x.b = b;
  union { unsigned u; float f; } y; y.u = ((unsigned)x.s) << 16;
  return y.f;
}

// 16-byte global -> LDS copy; async (no VGPR round trip) when available.
// Builtin signature (from hipcc diagnostic): (int4 addrspace(1)*, int4
// addrspace(3)*, imm offset, imm cpol).
__device__ __forceinline__ void copy16_g2l(const __bf16* __restrict__ g,
                                           unsigned short* __restrict__ l) {
#if USE_ASYNC_LDS
  __builtin_amdgcn_global_load_async_to_lds_b128(
      (__attribute__((address_space(1))) vint4*)g,
      (__attribute__((address_space(3))) vint4*)l, 0, 0);
#else
  *(uint4*)l = *(const uint4*)g;
#endif
}
__device__ __forceinline__ void async_wait_all() {
#if USE_ASYNC_LDS
  __builtin_amdgcn_s_wait_asynccnt(0);
#endif
}

// ---------------------------------------------------------------------------
// f32 -> bf16 conversion (weights, once per launch)
// ---------------------------------------------------------------------------
__global__ __launch_bounds__(256) void f32_to_bf16_kernel(
    const float* __restrict__ src, __bf16* __restrict__ dst, size_t n) {
  size_t i = (size_t)blockIdx.x * 256u + threadIdx.x;
  size_t st = (size_t)gridDim.x * 256u;
  for (; i < n; i += st) dst[i] = f2bf(src[i]);
}

// ---------------------------------------------------------------------------
// COMMIT masking + embedding:
//   masked = s<=0.15 ? (cond==0 ? 511 : cond) : tok
//   emb = token_emb[masked] + pos_enc + class_emb[task]
// ---------------------------------------------------------------------------
__global__ __launch_bounds__(256) void embed_kernel(
    const int* __restrict__ tok, const int* __restrict__ task,
    const int* __restrict__ cond, const float* __restrict__ sv,
    const float* __restrict__ temb, const float* __restrict__ pos,
    const float* __restrict__ cemb,
    float* __restrict__ ef, __bf16* __restrict__ eb) {
  int t = blockIdx.x;  // token 0..2047
  int b = t >> 9, si = t & 511;
  float s = sv[t];
  int tk = tok[t], cd = cond[t];
  int id = (s <= 0.15f) ? (cd == 0 ? 511 : cd) : tk;
  int ts = task[b];
#pragma unroll
  for (int i = 0; i < 3; i++) {
    int c = threadIdx.x + i * 256;
    float v = temb[(size_t)id * 768 + c] + pos[(size_t)si * 768 + c] +
              cemb[(size_t)ts * 768 + c];
    size_t idx = (size_t)t * 768 + c;
    ef[idx] = v;
    eb[idx] = f2bf(v);
  }
}

// ---------------------------------------------------------------------------
// Residual + LayerNorm over H=768: out = LN(x + res)*g + b, fp32 + bf16 out
// ---------------------------------------------------------------------------
__global__ __launch_bounds__(256) void ln_row_kernel(
    const float* __restrict__ x, const float* __restrict__ res,
    const float* __restrict__ g, const float* __restrict__ be,
    float* __restrict__ of, __bf16* __restrict__ ob) {
  __shared__ float red[256];
  size_t row = blockIdx.x;
  const float* xr = x + row * 768;
  const float* rr = res ? res + row * 768 : nullptr;
  int t = threadIdx.x;
  float v[3];
#pragma unroll
  for (int i = 0; i < 3; i++) {
    int c = t + i * 256;
    v[i] = xr[c] + (rr ? rr[c] : 0.0f);
  }
  red[t] = v[0] + v[1] + v[2];
  __syncthreads();
  for (int s = 128; s > 0; s >>= 1) { if (t < s) red[t] += red[t + s]; __syncthreads(); }
  float mean = red[0] * (1.0f / 768.0f);
  __syncthreads();
  float d0 = v[0] - mean, d1 = v[1] - mean, d2 = v[2] - mean;
  red[t] = d0 * d0 + d1 * d1 + d2 * d2;
  __syncthreads();
  for (int s = 128; s > 0; s >>= 1) { if (t < s) red[t] += red[t + s]; __syncthreads(); }
  float inv = rsqrtf(red[0] * (1.0f / 768.0f) + 1e-5f);
#pragma unroll
  for (int i = 0; i < 3; i++) {
    int c = t + i * 256;
    float o = (v[i] - mean) * inv * g[c] + be[c];
    size_t idx = row * 768 + c;
    of[idx] = o;
    ob[idx] = f2bf(o);
  }
}

// ---------------------------------------------------------------------------
// Row softmax over 512 bf16 scores, in place, with scale (1/sqrt(DH))
// ---------------------------------------------------------------------------
__global__ __launch_bounds__(256) void softmax_kernel(
    __bf16* __restrict__ p, float scale) {
  __shared__ float red[256];
  size_t row = blockIdx.x;
  __bf16* pr = p + row * 512;
  int t = threadIdx.x;
  float v0 = bf2f(pr[t]) * scale, v1 = bf2f(pr[t + 256]) * scale;
  red[t] = fmaxf(v0, v1);
  __syncthreads();
  for (int s = 128; s > 0; s >>= 1) { if (t < s) red[t] = fmaxf(red[t], red[t + s]); __syncthreads(); }
  float m = red[0];
  __syncthreads();
  float e0 = __expf(v0 - m), e1 = __expf(v1 - m);
  red[t] = e0 + e1;
  __syncthreads();
  for (int s = 128; s > 0; s >>= 1) { if (t < s) red[t] += red[t + s]; __syncthreads(); }
  float inv = 1.0f / red[0];
  pr[t] = f2bf(e0 * inv);
  pr[t + 256] = f2bf(e1 * inv);
}

// ---------------------------------------------------------------------------
// Generic tiled WMMA GEMM:  C[M,N] = A[M,K] * op(B) + bias
//   TRANSB=1: B stored N x K (weightlike, A @ B^T); async-staged.
//   TRANSB=0: B stored K x N (probs @ V); transposed on the LDS write.
// Block = 256 threads = 8 waves. Block tile 128(M) x 64(N), K stage 64,
// double-buffered LDS (ping-pong). Wave w owns rows [16w,16w+16) x 64 cols:
// 8 v_wmma_f32_16x16x32_bf16 per stage. Next stage's global->LDS traffic is
// issued after the first barrier so it overlaps with this stage's WMMAs;
// each wave drains its own ASYNCcnt before signaling the barrier.
// LDS pitch 72 halves (144B) keeps the ds_load_b128 fragment gathers 16B
// aligned and bank-conflict-free; gathers match the ISA 16-bit 16x32 (A) /
// 32x16 (B) wave32 layouts exactly.
// Batched via z: ptr += (z/innerCnt)*outerStride + (z%innerCnt)*innerStride.
// ---------------------------------------------------------------------------
template <bool TRANSB, bool RELU, bool WF, bool WB>
__global__ __launch_bounds__(256) void gemm_wmma(
    const __bf16* __restrict__ A, int lda, long aO, long aI, int aC,
    const __bf16* __restrict__ Bm, int ldb, long bO, long bI, int bC,
    float* __restrict__ Cf, __bf16* __restrict__ Cb, int ldc,
    long cO, long cI, int cC,
    const float* __restrict__ bias, int K) {
  __shared__ __align__(16) unsigned short As[2][128 * 72];  // 36864 B
  __shared__ __align__(16) unsigned short Bs[2][64 * 72];   // 18432 B

  int z = blockIdx.z;
  A  += (size_t)(z / aC) * aO + (size_t)(z % aC) * aI;
  Bm += (size_t)(z / bC) * bO + (size_t)(z % bC) * bI;
  size_t coff = (size_t)(z / cC) * cO + (size_t)(z % cC) * cI;
  float*  Cfp = nullptr;
  __bf16* Cbp = nullptr;
  if constexpr (WF) Cfp = Cf + coff;
  if constexpr (WB) Cbp = Cb + coff;

  int m0 = blockIdx.y * 128, n0 = blockIdx.x * 64;
  int tid = threadIdx.x;
  int w = tid >> 5, L = tid & 31;

  v8f acc[4] = {};

  // Stage one 64-wide K slab into LDS buffer sbuf.
  auto stage = [&](int k0, int sbuf) {
    // A tile: 128 rows x 64 k = 1024 16B chunks, 4 per thread.
#pragma unroll
    for (int j = 0; j < 4; j++) {
      int c = tid + j * 256;
      int row = c >> 3, col8 = c & 7;
      copy16_g2l(A + (size_t)(m0 + row) * lda + k0 + col8 * 8,
                 &As[sbuf][row * 72 + col8 * 8]);
    }
    if constexpr (TRANSB) {
      // B tile: 64 rows(n) x 64 k = 512 chunks, 2 per thread.
#pragma unroll
      for (int j = 0; j < 2; j++) {
        int c = tid + j * 256;
        int row = c >> 3, col8 = c & 7;
        copy16_g2l(Bm + (size_t)(n0 + row) * ldb + k0 + col8 * 8,
                   &Bs[sbuf][row * 72 + col8 * 8]);
      }
    } else {
      // B stored K x N: transpose into [n][k] on the LDS write.
#pragma unroll
      for (int j = 0; j < 2; j++) {
        int c = tid + j * 256;
        int kk = c >> 3, n8 = c & 7;
        const uint4* src =
            (const uint4*)(Bm + (size_t)(k0 + kk) * ldb + n0 + n8 * 8);
        unsigned short tmp[8];
        *(uint4*)tmp = *src;
#pragma unroll
        for (int q = 0; q < 8; q++) Bs[sbuf][(n8 * 8 + q) * 72 + kk] = tmp[q];
      }
    }
  };

  int nk = K >> 6;  // all K here are multiples of 64
  stage(0, 0);
  for (int i = 0; i < nk; i++) {
    int cur = i & 1;
    async_wait_all();    // own async writes for buf[cur] complete
    __syncthreads();     // buf[cur] visible to all waves
    if (i + 1 < nk) stage((i + 1) << 6, cur ^ 1);  // overlap with WMMAs below

#pragma unroll
    for (int ks = 0; ks < 2; ks++) {
      const __bf16* ap = (const __bf16*)&As[cur][(w * 16 + (L & 15)) * 72 +
                                                 ks * 32 + ((L >> 4) * 8)];
      v8bf alo = *(const v8bf*)ap;
      v8bf ahi = *(const v8bf*)(ap + 16);
      v16bf a = __builtin_shufflevector(alo, ahi, 0, 1, 2, 3, 4, 5, 6, 7, 8, 9,
                                        10, 11, 12, 13, 14, 15);
#pragma unroll
      for (int t = 0; t < 4; t++) {
        const __bf16* bp = (const __bf16*)&Bs[cur][((L & 15) + 16 * t) * 72 +
                                                   ks * 32 + ((L >> 4) * 8)];
        v8bf blo = *(const v8bf*)bp;
        v8bf bhi = *(const v8bf*)(bp + 16);
        v16bf b = __builtin_shufflevector(blo, bhi, 0, 1, 2, 3, 4, 5, 6, 7, 8,
                                          9, 10, 11, 12, 13, 14, 15);
        acc[t] = __builtin_amdgcn_wmma_f32_16x16x32_bf16(
            false, a, false, b, (short)0, acc[t], false, false);
      }
    }
    __syncthreads();     // all reads of buf[cur] done before it is reused
  }

  // ---- epilogue: C/D layout — VGPR r: M = r + (lane>=16 ? 8:0), N = lane&15
  int mrel = w * 16 + ((L >> 4) * 8);
  int colL = L & 15;
#pragma unroll
  for (int t = 0; t < 4; t++) {
    int n = n0 + t * 16 + colL;
    float bv = bias ? bias[n] : 0.0f;
#pragma unroll
    for (int r = 0; r < 8; r++) {
      float v = acc[t][r] + bv;
      if (RELU) v = v > 0.0f ? v : 0.0f;
      size_t idx = (size_t)(m0 + mrel + r) * ldc + n;
      if constexpr (WF) Cfp[idx] = v;
      if constexpr (WB) Cbp[idx] = f2bf(v);
    }
  }
}

#define LAUNCH_GEMM(TB, RL, WFf, WBf, M, N, Kk, BT, A, lda, aO, aI, aC, Bp,   \
                    ldb, bO, bI, bC, Cfp, Cbp, ldc, cO, cI, cC, biasp)        \
  gemm_wmma<TB, RL, WFf, WBf><<<dim3((N) / 64, (M) / 128, (BT)), 256, 0,      \
                                stream>>>(                                    \
      A, lda, aO, aI, aC, Bp, ldb, bO, bI, bC, Cfp, Cbp, ldc, cO, cI, cC,     \
      biasp, Kk)

// ---------------------------------------------------------------------------
// Host orchestration
// ---------------------------------------------------------------------------
extern "C" void kernel_launch(void* const* d_in, const int* in_sizes, int n_in,
                              void* d_out, int out_size, void* d_ws,
                              size_t ws_size, hipStream_t stream) {
  (void)in_sizes; (void)n_in; (void)out_size; (void)ws_size;

  const int B = 4, S = 512, H = 768, NH = 12, DH = 64, DFF = 2048, L = 6,
            V = 512;
  const int T = B * S, H3 = 3 * H, H2 = 2 * H;
  const long SH3 = (long)S * H3, SS = (long)S * S, SH = (long)S * H;

  // ---- inputs: top-level in setup_inputs() insertion order; nested param
  // dicts flattened with sorted keys (jax pytree convention) ----
  const int*   tokens    = (const int*)d_in[0];
  const int*   task_ids  = (const int*)d_in[1];
  const int*   cond_tok  = (const int*)d_in[2];
  const float* s_vals    = (const float*)d_in[3];
  const float* token_emb = (const float*)d_in[4];
  const float* pos_enc   = (const float*)d_in[5];
  const float* class_emb = (const float*)d_in[6];
  const float* out_w     = (const float*)d_in[7];
  const float* out_b     = (const float*)d_in[8];
  const float* e_ff1_b = (const float*)d_in[9];
  const float* e_ff1_w = (const float*)d_in[10];
  const float* e_ff2_b = (const float*)d_in[11];
  const float* e_ff2_w = (const float*)d_in[12];
  const float* e_ln1_b = (const float*)d_in[13];
  const float* e_ln1_s = (const float*)d_in[14];
  const float* e_ln2_b = (const float*)d_in[15];
  const float* e_ln2_s = (const float*)d_in[16];
  const float* e_o_b   = (const float*)d_in[17];
  const float* e_o_w   = (const float*)d_in[18];
  const float* e_qkv_b = (const float*)d_in[19];
  const float* e_qkv_w = (const float*)d_in[20];
  const float* e_nrm_b = (const float*)d_in[21];
  const float* e_nrm_s = (const float*)d_in[22];
  const float* d_cao_b  = (const float*)d_in[23];
  const float* d_cao_w  = (const float*)d_in[24];
  const float* d_caq_b  = (const float*)d_in[25];
  const float* d_caq_w  = (const float*)d_in[26];
  const float* d_ff1_b  = (const float*)d_in[27];
  const float* d_ff1_w  = (const float*)d_in[28];
  const float* d_ff2_b  = (const float*)d_in[29];
  const float* d_ff2_w  = (const float*)d_in[30];
  const float* d_ln1_b  = (const float*)d_in[31];
  const float* d_ln1_s  = (const float*)d_in[32];
  const float* d_ln2_b  = (const float*)d_in[33];
  const float* d_ln2_s  = (const float*)d_in[34];
  const float* d_ln3_b  = (const float*)d_in[35];
  const float* d_ln3_s  = (const float*)d_in[36];
  const float* d_o_b    = (const float*)d_in[37];
  const float* d_o_w    = (const float*)d_in[38];
  const float* d_qkv_b  = (const float*)d_in[39];
  const float* d_qkv_w  = (const float*)d_in[40];
  const float* d_nrm_b  = (const float*)d_in[41];
  const float* d_nrm_s  = (const float*)d_in[42];

  // ---- workspace bump allocator ----
  char* ws = (char*)d_ws;
  size_t off = 0;
  auto allocB = [&](size_t elems) {
    __bf16* p = (__bf16*)(ws + off);
    off += ((elems * 2 + 255) / 256) * 256;
    return p;
  };
  auto allocF = [&](size_t elems) {
    float* p = (float*)(ws + off);
    off += ((elems * 4 + 255) / 256) * 256;
    return p;
  };

  __bf16* wb_eqkv = allocB((size_t)L * H3 * H);
  __bf16* wb_eo   = allocB((size_t)L * H * H);
  __bf16* wb_ef1  = allocB((size_t)L * DFF * H);
  __bf16* wb_ef2  = allocB((size_t)L * H * DFF);
  __bf16* wb_dqkv = allocB((size_t)L * H3 * H);
  __bf16* wb_do   = allocB((size_t)L * H * H);
  __bf16* wb_df1  = allocB((size_t)L * DFF * H);
  __bf16* wb_df2  = allocB((size_t)L * H * DFF);
  __bf16* wb_dcaq = allocB((size_t)L * H3 * H);
  __bf16* wb_dcao = allocB((size_t)L * H * H);
  __bf16* wb_out  = allocB((size_t)V * H);

  __bf16* emb_b = allocB((size_t)T * H);
  __bf16* hb    = allocB((size_t)T * H);
  __bf16* memb  = allocB((size_t)T * H);
  __bf16* qkvb  = allocB((size_t)T * H3);
  __bf16* probs = allocB((size_t)B * NH * S * S);
  __bf16* attnb = allocB((size_t)T * H);
  __bf16* ffn1b = allocB((size_t)T * DFF);

  float* emb_f = allocF((size_t)T * H);
  float* h_f   = allocF((size_t)T * H);
  float* oproj = allocF((size_t)T * H);
  float* ffn2f = allocF((size_t)T * H);

  // ---- weight conversion (per launch; deterministic) ----
  auto conv = [&](const float* s, __bf16* d, size_t n) {
    f32_to_bf16_kernel<<<2048, 256, 0, stream>>>(s, d, n);
  };
  conv(e_qkv_w, wb_eqkv, (size_t)L * H3 * H);
  conv(e_o_w,   wb_eo,   (size_t)L * H * H);
  conv(e_ff1_w, wb_ef1,  (size_t)L * DFF * H);
  conv(e_ff2_w, wb_ef2,  (size_t)L * H * DFF);
  conv(d_qkv_w, wb_dqkv, (size_t)L * H3 * H);
  conv(d_o_w,   wb_do,   (size_t)L * H * H);
  conv(d_ff1_w, wb_df1,  (size_t)L * DFF * H);
  conv(d_ff2_w, wb_df2,  (size_t)L * H * DFF);
  conv(d_caq_w, wb_dcaq, (size_t)L * H3 * H);
  conv(d_cao_w, wb_dcao, (size_t)L * H * H);
  conv(out_w,   wb_out,  (size_t)V * H);

  // ---- embedding ----
  embed_kernel<<<T, 256, 0, stream>>>(tokens, task_ids, cond_tok, s_vals,
                                      token_emb, pos_enc, class_emb, emb_f,
                                      emb_b);

  const float attn_scale = 0.125f;  // 1/sqrt(64)

  // scores -> softmax -> probs @ V, batched over (b, head)
  auto attention_core = [&]() {
    LAUNCH_GEMM(true, false, false, true, S, S, DH, B * NH,
                qkvb, H3, SH3, (long)DH, NH,
                qkvb + H, H3, SH3, (long)DH, NH,
                (float*)nullptr, probs, S, SS, 0L, 1, (const float*)nullptr);
    softmax_kernel<<<B * NH * S, 256, 0, stream>>>(probs, attn_scale);
    LAUNCH_GEMM(false, false, false, true, S, DH, S, B * NH,
                probs, S, SS, 0L, 1,
                qkvb + H2, H3, SH3, (long)DH, NH,
                (float*)nullptr, attnb, H, SH, (long)DH, NH,
                (const float*)nullptr);
  };

  // ======================= ENCODER =======================
  const float* in_f = emb_f;
  const __bf16* in_b = emb_b;
  for (int l = 0; l < L; l++) {
    size_t qw = (size_t)l * H3 * H, ow = (size_t)l * H * H;
    size_t f1 = (size_t)l * DFF * H, f2 = (size_t)l * H * DFF;
    LAUNCH_GEMM(true, false, false, true, T, H3, H, 1,
                in_b, H, 0L, 0L, 1, wb_eqkv + qw, H, 0L, 0L, 1,
                (float*)nullptr, qkvb, H3, 0L, 0L, 1,
                e_qkv_b + (size_t)l * H3);
    attention_core();
    LAUNCH_GEMM(true, false, true, false, T, H, H, 1,
                attnb, H, 0L, 0L, 1, wb_eo + ow, H, 0L, 0L, 1,
                oproj, (__bf16*)nullptr, H, 0L, 0L, 1, e_o_b + (size_t)l * H);
    ln_row_kernel<<<T, 256, 0, stream>>>(oproj, in_f, e_ln1_s + (size_t)l * H,
                                         e_ln1_b + (size_t)l * H, h_f, hb);
    LAUNCH_GEMM(true, true, false, true, T, DFF, H, 1,
                hb, H, 0L, 0L, 1, wb_ef1 + f1, H, 0L, 0L, 1,
                (float*)nullptr, ffn1b, DFF, 0L, 0L, 1,
                e_ff1_b + (size_t)l * DFF);
    LAUNCH_GEMM(true, false, true, false, T, H, DFF, 1,
                ffn1b, DFF, 0L, 0L, 1, wb_ef2 + f2, DFF, 0L, 0L, 1,
                ffn2f, (__bf16*)nullptr, H, 0L, 0L, 1, e_ff2_b + (size_t)l * H);
    ln_row_kernel<<<T, 256, 0, stream>>>(ffn2f, h_f, e_ln2_s + (size_t)l * H,
                                         e_ln2_b + (size_t)l * H, h_f, hb);
    in_f = h_f;
    in_b = hb;
  }
  ln_row_kernel<<<T, 256, 0, stream>>>(h_f, (const float*)nullptr, e_nrm_s,
                                       e_nrm_b, h_f, memb);

  // ======================= DECODER =======================
  in_f = emb_f;
  in_b = emb_b;
  for (int l = 0; l < L; l++) {
    size_t qw = (size_t)l * H3 * H, ow = (size_t)l * H * H;
    size_t f1 = (size_t)l * DFF * H, f2 = (size_t)l * H * DFF;
    // self attention
    LAUNCH_GEMM(true, false, false, true, T, H3, H, 1,
                in_b, H, 0L, 0L, 1, wb_dqkv + qw, H, 0L, 0L, 1,
                (float*)nullptr, qkvb, H3, 0L, 0L, 1,
                d_qkv_b + (size_t)l * H3);
    attention_core();
    LAUNCH_GEMM(true, false, true, false, T, H, H, 1,
                attnb, H, 0L, 0L, 1, wb_do + ow, H, 0L, 0L, 1,
                oproj, (__bf16*)nullptr, H, 0L, 0L, 1, d_o_b + (size_t)l * H);
    ln_row_kernel<<<T, 256, 0, stream>>>(oproj, in_f, d_ln1_s + (size_t)l * H,
                                         d_ln1_b + (size_t)l * H, h_f, hb);
    // cross attention: Q from h, K/V from encoder memory
    LAUNCH_GEMM(true, false, false, true, T, H, H, 1,
                hb, H, 0L, 0L, 1, wb_dcaq + qw, H, 0L, 0L, 1,
                (float*)nullptr, qkvb, H3, 0L, 0L, 1,
                d_caq_b + (size_t)l * H3);
    LAUNCH_GEMM(true, false, false, true, T, H2, H, 1,
                memb, H, 0L, 0L, 1, wb_dcaq + qw + (size_t)H * H, H, 0L, 0L, 1,
                (float*)nullptr, qkvb + H, H3, 0L, 0L, 1,
                d_caq_b + (size_t)l * H3 + H);
    attention_core();
    LAUNCH_GEMM(true, false, true, false, T, H, H, 1,
                attnb, H, 0L, 0L, 1, wb_dcao + ow, H, 0L, 0L, 1,
                oproj, (__bf16*)nullptr, H, 0L, 0L, 1, d_cao_b + (size_t)l * H);
    ln_row_kernel<<<T, 256, 0, stream>>>(oproj, h_f, d_ln2_s + (size_t)l * H,
                                         d_ln2_b + (size_t)l * H, h_f, hb);
    // FFN
    LAUNCH_GEMM(true, true, false, true, T, DFF, H, 1,
                hb, H, 0L, 0L, 1, wb_df1 + f1, H, 0L, 0L, 1,
                (float*)nullptr, ffn1b, DFF, 0L, 0L, 1,
                d_ff1_b + (size_t)l * DFF);
    LAUNCH_GEMM(true, false, true, false, T, H, DFF, 1,
                ffn1b, DFF, 0L, 0L, 1, wb_df2 + f2, DFF, 0L, 0L, 1,
                ffn2f, (__bf16*)nullptr, H, 0L, 0L, 1, d_ff2_b + (size_t)l * H);
    ln_row_kernel<<<T, 256, 0, stream>>>(ffn2f, h_f, d_ln3_s + (size_t)l * H,
                                         d_ln3_b + (size_t)l * H, h_f, hb);
    in_f = h_f;
    in_b = hb;
  }
  ln_row_kernel<<<T, 256, 0, stream>>>(h_f, (const float*)nullptr, d_nrm_s,
                                       d_nrm_b, h_f, hb);

  // output projection: [T,768] @ out_w^T[768,512] + out_b -> d_out fp32
  LAUNCH_GEMM(true, false, true, false, T, V, H, 1,
              hb, H, 0L, 0L, 1, wb_out, H, 0L, 0L, 1,
              (float*)d_out, (__bf16*)nullptr, V, 0L, 0L, 1, out_b);
}
